// _ScaledDotProduct_4105988735943
// MI455X (gfx1250) — matrix operand order repaired and checked by reading, not compiled, not run on hardware
//
#include <hip/hip_runtime.h>
#include <hip/hip_bf16.h>

typedef __attribute__((ext_vector_type(16))) _Float16 v16h;
typedef __attribute__((ext_vector_type(8)))  float    v8f;

#define S_LEN 2048
#define HD    64
#define SCALE 0.125f
#define TM    16
#define SSTR  2052      // padded LDS row stride (floats) to break bank conflicts

__global__ __launch_bounds__(256)
void _ScaledDotProduct_attn_kernel(const float* __restrict__ q,
                                   const float* __restrict__ k,
                                   const float* __restrict__ v,
                                   const float* __restrict__ prev,
                                   float* __restrict__ out,      // B*H*S*HD
                                   float* __restrict__ scores)   // B*H*S*S
{
    __shared__ _Float16 qs[TM * HD];        // 2 KB
    __shared__ float    sc[TM * SSTR];      // 128.25 KB score tile
    __shared__ float    red[16 * 16];
    __shared__ float    rowmax[16];
    __shared__ float    rowinv[16];
    __shared__ float    pb[TM * HD];        // 4 KB split-K partials

    const int tid  = threadIdx.x;
    const int lane = tid & 31;
    const int wave = tid >> 5;
    const int hi   = (lane >= 16) ? 1 : 0;
    const int ln   = lane & 15;

    const int bh    = blockIdx.x >> 7;      // / (S/16 = 128)
    const int mtile = blockIdx.x & 127;
    const int m0    = mtile * TM;

    const float* qp = q    + (size_t)bh * S_LEN * HD + (size_t)m0 * HD;
    const float* kp = k    + (size_t)bh * HD * S_LEN;               // (d, S) layout
    const float* vp = v    + (size_t)bh * S_LEN * HD;               // (S, d) layout
    const float* pp = prev + (size_t)bh * S_LEN * S_LEN + (size_t)m0 * S_LEN;
    float* op = out    + (size_t)bh * S_LEN * HD + (size_t)m0 * HD;
    float* sp = scores + (size_t)bh * S_LEN * S_LEN + (size_t)m0 * S_LEN;

    // ---- Phase 1: Q tile (16x64 f32) * SCALE -> LDS f16 (exact: 2^-3) ----
    for (int i = tid; i < TM * HD; i += 256)
        qs[i] = (_Float16)(qp[i] * SCALE);
    __syncthreads();

    // A-fragments for Q (16x32 f16 layout, ISA 7.12.2): row m = ln,
    // half h -> K = (j<4 ? 2j : 16+2(j-4)) + 8*hi + (h&1), j = h>>1
    v16h a0, a1;
    #pragma unroll
    for (int h = 0; h < 16; ++h) {
        int jj   = h >> 1;
        int koff = ((jj < 4) ? (2 * jj) : (16 + 2 * (jj - 4))) + (hi ? 8 : 0) + (h & 1);
        a0[h] = qs[ln * HD + koff];
        a1[h] = qs[ln * HD + 32 + koff];
    }

    // ---- Phase 2: scores = (Q*scale) @ K + prev ; write global + LDS ----
    for (int cb = 0; cb < 16; ++cb) {
        const int jblk = wave * 16 + cb;        // column block 0..127
        const int n0   = jblk * 16;
        // B-fragments from K (K stored (d,S): row dd contiguous over t)
        // B 32x16 f16 layout: col n = ln, half h -> K = (h<8 ? h : h+8) + 8*hi
        v16h b0, b1;
        #pragma unroll
        for (int h = 0; h < 16; ++h) {
            int kk = ((h < 8) ? h : (h + 8)) + (hi ? 8 : 0);
            b0[h] = (_Float16)kp[(size_t)kk * S_LEN + n0 + ln];
            b1[h] = (_Float16)kp[(size_t)(kk + 32) * S_LEN + n0 + ln];
        }
        v8f c = {};
        c = __builtin_amdgcn_wmma_f32_16x16x32_f16(false, a0, false, b0, (short)0, c, false, false);
        c = __builtin_amdgcn_wmma_f32_16x16x32_f16(false, a1, false, b1, (short)0, c, false, false);
        // C layout: VGPR r -> row m = r + 8*hi, col n = ln
        #pragma unroll
        for (int r = 0; r < 8; ++r) {
            int   m    = r + hi * 8;
            float sval = c[r] + pp[(size_t)m * S_LEN + n0 + ln];
            sp[(size_t)m * S_LEN + n0 + ln] = sval;   // pre-softmax scores out
            sc[m * SSTR + n0 + ln]          = sval;   // keep tile for softmax
        }
    }
    __syncthreads();

    // ---- Phase 3: row softmax over LDS tile (deterministic, no atomics) ----
    {
        const int row = tid >> 4;     // 16 threads per row
        const int c0  = tid & 15;
        float mx = -3.0e38f;
        for (int i = 0; i < S_LEN / 16; ++i)
            mx = fmaxf(mx, sc[row * SSTR + c0 + 16 * i]);
        red[row * 16 + c0] = mx;
        __syncthreads();
        if (c0 == 0) {
            float m2 = red[row * 16];
            for (int i = 1; i < 16; ++i) m2 = fmaxf(m2, red[row * 16 + i]);
            rowmax[row] = m2;
        }
        __syncthreads();
        const float rm = rowmax[row];
        float sum = 0.0f;
        for (int i = 0; i < S_LEN / 16; ++i) {
            int   idx = row * SSTR + c0 + 16 * i;
            float e   = __expf(sc[idx] - rm);
            sc[idx]   = e;                 // overwrite with exp values
            sum += e;
        }
        red[row * 16 + c0] = sum;
        __syncthreads();
        if (c0 == 0) {
            float s2 = 0.0f;
            for (int i = 0; i < 16; ++i) s2 += red[row * 16 + i];
            rowinv[row] = 1.0f / s2;
        }
    }
    __syncthreads();

    // ---- Phase 4: out = softmax(scores) @ V, split-K across all 8 waves ----
    // waves 0..3: N-block nb over k-blocks 0..31; waves 4..7: same nb, 32..63
    {
        const int   nb  = wave & 3;        // output column block (16 of HD=64)
        const int   kh  = wave >> 2;       // K half
        const float inv = rowinv[ln];      // A-fragment row for this lane
        v8f o = {};
        for (int kb = kh * 32; kb < kh * 32 + 32; ++kb) {
            const int kbase = kb * 32;
            v16h aw;                        // weights fragment (f32 LDS -> f16)
            #pragma unroll
            for (int h = 0; h < 16; ++h) {
                int jj   = h >> 1;
                int koff = ((jj < 4) ? (2 * jj) : (16 + 2 * (jj - 4))) + (hi ? 8 : 0) + (h & 1);
                aw[h] = (_Float16)(sc[ln * SSTR + kbase + koff] * inv);
            }
            v16h bv;                        // V fragment: (S,d) layout, row t contiguous
            #pragma unroll
            for (int h = 0; h < 16; ++h) {
                int t = kbase + ((h < 8) ? h : (h + 8)) + (hi ? 8 : 0);
                bv[h] = (_Float16)vp[(size_t)t * HD + nb * 16 + ln];
            }
            o = __builtin_amdgcn_wmma_f32_16x16x32_f16(false, aw, false, bv, (short)0, o, false, false);
        }
        // Upper K-half waves spill partials to LDS
        if (wave >= 4) {
            #pragma unroll
            for (int r = 0; r < 8; ++r) {
                int m = r + hi * 8;
                pb[m * HD + nb * 16 + ln] = o[r];
            }
        }
        __syncthreads();
        // Lower waves combine (fixed order: lower + upper -> deterministic)
        if (wave < 4) {
            #pragma unroll
            for (int r = 0; r < 8; ++r) {
                int m = r + hi * 8;
                op[(size_t)m * HD + nb * 16 + ln] = o[r] + pb[m * HD + nb * 16 + ln];
            }
        }
    }
}

extern "C" void kernel_launch(void* const* d_in, const int* in_sizes, int n_in,
                              void* d_out, int out_size, void* d_ws, size_t ws_size,
                              hipStream_t stream) {
    const float* q    = (const float*)d_in[0];
    const float* k    = (const float*)d_in[1];
    const float* v    = (const float*)d_in[2];
    const float* prev = (const float*)d_in[3];

    const size_t OUT_ELEMS = (size_t)2 * 16 * 2048 * 64;   // B*H*S*d
    float* out    = (float*)d_out;
    float* scores = (float*)d_out + OUT_ELEMS;

    dim3 grid(2 * 16 * (S_LEN / TM));   // 4096 workgroups
    _ScaledDotProduct_attn_kernel<<<grid, 256, 0, stream>>>(q, k, v, prev, out, scores);
}
